// E8P12RVQ4B_codebook_26877905339289
// MI455X (gfx1250) — matrix operand order repaired and checked by reading, not compiled, not run on hardware
//
#include <hip/hip_runtime.h>

// ---------------------------------------------------------------------------
// E8P residual-VQ dequant + GEMM for MI455X (gfx1250, wave32, WMMA).
//   out[M,N] = input[M,K] @ W[N,K]^T,  W = grid[hi] + (1/3.45)*grid[lo]
// Compute-bound (AI ~1000 FLOP/B) -> use v_wmma_f32_16x16x32_bf16.
// ---------------------------------------------------------------------------

typedef __bf16 bf16_t;
typedef __attribute__((ext_vector_type(16))) bf16_t v16bf;
typedef __attribute__((ext_vector_type(8)))  bf16_t v8bf;
typedef __attribute__((ext_vector_type(8)))  float  v8f;
typedef __attribute__((ext_vector_type(4)))  float  v4f;
typedef __attribute__((ext_vector_type(8)))  unsigned short us8;
typedef __attribute__((ext_vector_type(4)))  unsigned short us4;
typedef __attribute__((ext_vector_type(4)))  int    i4;

constexpr int   Mdim   = 8192;
constexpr int   Ndim   = 4096;
constexpr int   Kdim   = 4096;
constexpr float RSCALE = 1.0f / 3.45f;

constexpr int BM   = 128;       // block tile M
constexpr int BN   = 128;       // block tile N
constexpr int BK   = 64;        // block tile K
constexpr int LDSA = BK + 8;    // ushort stride (144B rows, 16B aligned, pad vs banks)
constexpr int LDSB = BK + 8;

// float -> bf16, round-to-nearest-even
__device__ __forceinline__ unsigned short f2bf(float f) {
  unsigned u = __float_as_uint(f);
  u += 0x7FFFu + ((u >> 16) & 1u);
  return (unsigned short)(u >> 16);
}

union V16U {
  struct { v8bf lo, hi; } p;
  v16bf v;
};

__global__ __launch_bounds__(256)
void e8p_rvq_gemm(const float* __restrict__ inp,
                  const int*   __restrict__ qidx,
                  const float* __restrict__ grid,
                  float*       __restrict__ out) {
  __shared__ unsigned short As[BM * LDSA];   // A tile, bf16 bits
  __shared__ unsigned short Bs[BN * LDSB];   // dequantized W tile, bf16 bits

  const int tid = threadIdx.x;
  const int n0  = blockIdx.x * BN;
  const int m0  = blockIdx.y * BM;

  const int lane   = tid & 31;
  const int wave   = tid >> 5;
  const int lane16 = lane & 15;
  const int half   = lane >> 4;
  const int wm     = (wave & 3) * 32;   // wave's M offset inside tile (4 waves)
  const int wn     = (wave >> 2) * 64;  // wave's N offset inside tile (2 waves)

  v8f acc[2][4];
  const v8f vzero = {};
#pragma unroll
  for (int i = 0; i < 2; ++i)
#pragma unroll
    for (int j = 0; j < 4; ++j) acc[i][j] = vzero;

  // dequant work split: thread -> (row in tile, group of 4 codes)
  const int drow  = tid >> 1;   // 0..127
  const int dpair = tid & 1;    // 0..1  (codes 0..3 or 4..7 of this BK slice)

  for (int kk = 0; kk < Kdim; kk += BK) {
    // ---- stage A: 128x64 f32 -> bf16 into LDS ----
#pragma unroll
    for (int i = 0; i < 8; ++i) {
      const int f  = tid + i * 256;     // float4 index within tile
      const int r  = f >> 4;
      const int c4 = f & 15;
      v4f v = *(const v4f*)(inp + (size_t)(m0 + r) * Kdim + kk + c4 * 4);
      us4 w;
      w[0] = f2bf(v[0]); w[1] = f2bf(v[1]); w[2] = f2bf(v[2]); w[3] = f2bf(v[3]);
      *(us4*)&As[r * LDSA + c4 * 4] = w;
    }

    // ---- stage B: dequantize 128x64 weight tile into LDS ----
    {
      const i4 q4 = *(const i4*)(qidx + (size_t)(n0 + drow) * (Kdim / 8)
                                 + (kk >> 3) + dpair * 4);
#pragma unroll
      for (int j = 0; j < 4; ++j) {
        const unsigned q  = (unsigned)q4[j];
        const v4f* gh = (const v4f*)(grid + (size_t)(q >> 16) * 8);
        const v4f* gl = (const v4f*)(grid + (size_t)(q & 0xFFFFu) * 8);
        const v4f h0 = gh[0], h1 = gh[1];
        const v4f l0 = gl[0], l1 = gl[1];
        us8 w;
#pragma unroll
        for (int e = 0; e < 4; ++e) {
          w[e]     = f2bf(fmaf(RSCALE, l0[e], h0[e]));
          w[e + 4] = f2bf(fmaf(RSCALE, l1[e], h1[e]));
        }
        *(us8*)&Bs[drow * LDSB + dpair * 32 + j * 8] = w;
      }
    }

    // prefetch next A tile slice (gfx1250 global_prefetch_b8; speculative-safe)
    if (kk + BK < Kdim)
      __builtin_prefetch(inp + (size_t)(m0 + drow) * Kdim + kk + BK + dpair * 32, 0, 1);

    __syncthreads();

    // ---- compute: 2 k-chunks of 32, 2x4 WMMA tiles per wave ----
#pragma unroll
    for (int kc = 0; kc < 2; ++kc) {
      v16bf afrag[2], bfrag[4];

#pragma unroll
      for (int tm = 0; tm < 2; ++tm) {
        // A 16x32 bf16 layout: lane row M=lane16; K runs {8*half..+8} and {16+8*half..+8}
        const int base = (wm + tm * 16 + lane16) * LDSA + kc * 32 + half * 8;
        V16U u;
        u.p.lo = *(const v8bf*)&As[base];
        u.p.hi = *(const v8bf*)&As[base + 16];
        afrag[tm] = u.v;
      }
#pragma unroll
      for (int tn = 0; tn < 4; ++tn) {
        // B 32x16 bf16 layout: lane column N=lane16; K runs {8*half..+8} and {16+8*half..+8}
        const int base = (wn + tn * 16 + lane16) * LDSB + kc * 32 + half * 8;
        V16U u;
        u.p.lo = *(const v8bf*)&Bs[base];
        u.p.hi = *(const v8bf*)&Bs[base + 16];
        bfrag[tn] = u.v;
      }

#pragma unroll
      for (int tm = 0; tm < 2; ++tm)
#pragma unroll
        for (int tn = 0; tn < 4; ++tn)
          acc[tm][tn] = __builtin_amdgcn_wmma_f32_16x16x32_bf16(
              false, afrag[tm], false, bfrag[tn],
              (short)0, acc[tm][tn], false, false);
    }
    __syncthreads();
  }

  // ---- writeback: C/D layout = VGPR r -> row (r + 8*half), col lane16 ----
#pragma unroll
  for (int tm = 0; tm < 2; ++tm) {
#pragma unroll
    for (int tn = 0; tn < 4; ++tn) {
      const int row0 = m0 + wm + tm * 16 + 8 * half;
      const int col  = n0 + wn + tn * 16 + lane16;
      float* op = out + (size_t)row0 * Ndim + col;
#pragma unroll
      for (int r = 0; r < 8; ++r)
        op[(size_t)r * Ndim] = acc[tm][tn][r];
    }
  }
}

extern "C" void kernel_launch(void* const* d_in, const int* in_sizes, int n_in,
                              void* d_out, int out_size, void* d_ws, size_t ws_size,
                              hipStream_t stream) {
  (void)in_sizes; (void)n_in; (void)out_size; (void)d_ws; (void)ws_size;
  const float* inp  = (const float*)d_in[0];
  const int*   qidx = (const int*)d_in[1];
  const float* grid = (const float*)d_in[2];
  float* out = (float*)d_out;

  dim3 grd(Ndim / BN, Mdim / BM);   // 32 x 64 blocks
  e8p_rvq_gemm<<<grd, 256, 0, stream>>>(inp, qidx, grid, out);
}
